// AttentionLayer_8383776162534
// MI455X (gfx1250) — matrix operand order repaired and checked by reading, not compiled, not run on hardware
//
#include <hip/hip_runtime.h>
#include <cmath>

// ---------------- problem constants ----------------
constexpr int kDM  = 2048;           // d_model
constexpr int kH   = 16;             // query heads
constexpr int kHKV = 4;              // kv heads
constexpr int kDS  = 32;             // semantic dim
constexpr int kDG  = 32;             // geometric dim
constexpr int kDV  = 64;             // value dim
constexpr int kB   = 2;
constexpr int kT   = 4096;
constexpr int kMB  = 64;             // mem block
constexpr int kLW  = 1024;           // local window
constexpr int kRem = kT - kLW;       // 3072
constexpr int kNB  = kRem / kMB;     // 48 pooled blocks
constexpr int kKT  = kNB + kLW;      // 1072 real keys
constexpr int kKP  = 1088;           // padded to 34*32
constexpr int kBT  = kB * kT;        // 8192 token rows

typedef unsigned short u16;
typedef __bf16 bhalf;
typedef bhalf v16bf __attribute__((ext_vector_type(16)));
typedef float v8f   __attribute__((ext_vector_type(8)));
typedef u16  v8u    __attribute__((ext_vector_type(8)));
typedef u16  v16u   __attribute__((ext_vector_type(16)));

__device__ __forceinline__ u16 f2bf(float f) {
  union { float f; unsigned u; } v; v.f = f;
  return (u16)((v.u + 0x7FFFu + ((v.u >> 16) & 1u)) >> 16);  // RNE
}

// A-operand (16x32, 16-bit): per lane 16 consecutive K elems
__device__ __forceinline__ v16bf ld_a(const u16* p) {
  v8u lo = *(const v8u*)p;
  v8u hi = *(const v8u*)(p + 8);
  v16u t;
#pragma unroll
  for (int i = 0; i < 8; ++i) { t[i] = lo[i]; t[i + 8] = hi[i]; }
  return __builtin_bit_cast(v16bf, t);
}
// B-operand (32x16, 16-bit): per lane 16 consecutive K elems
__device__ __forceinline__ v16bf ld_b(const u16* p) {
  v8u lo = *(const v8u*)p;
  v8u hi = *(const v8u*)(p + 8);
  v16u t;
#pragma unroll
  for (int i = 0; i < 8; ++i) { t[i] = lo[i]; t[i + 8] = hi[i]; }
  return __builtin_bit_cast(v16bf, t);
}

#define WMMA_BF16(A, Bv, Cv) \
  __builtin_amdgcn_wmma_f32_16x16x32_bf16(false, (A), false, (Bv), (short)0, (Cv), false, false)

// ---------------- elementwise prep kernels ----------------
__global__ void k_cvt_bf16(const float* __restrict__ s, u16* __restrict__ d, int n) {
  int i = blockIdx.x * 256 + threadIdx.x;
  if (i < n) d[i] = f2bf(s[i]);
}

// src f32 [K,N] row-major -> dst bf16 [N,K] row-major
__global__ void k_transpose_bf16(const float* __restrict__ s, u16* __restrict__ d,
                                 int K, int N) {
  int i = blockIdx.x * 256 + threadIdx.x;
  if (i >= K * N) return;
  int k = i / N, n = i - k * N;
  d[(size_t)n * K + k] = f2bf(s[i]);
}

// ---------------- generic bf16 WMMA GEMM:  C[M,N] f32 = A[M,K] * Bt[N,K]^T ----------------
// One wave computes a 32x64 tile: 2 A-frags x 4 B-frags -> 8 WMMAs per K-step of 32.
__global__ void __launch_bounds__(256)
k_gemm_bf16(const u16* __restrict__ A, const u16* __restrict__ Bt,
            float* __restrict__ C, int M, int N, int K) {
  int wid  = blockIdx.x * 8 + (threadIdx.x >> 5);
  int lane = threadIdx.x & 31;
  int tiles_n = N >> 6;
  int tm = wid / tiles_n;
  int tn = wid - tm * tiles_n;
  if (tm * 32 >= M) return;                 // wave-uniform; EXEC stays full for WMMA
  int hl  = lane >> 4;
  int l15 = lane & 15;
  const u16* ap0 = A + (size_t)(tm * 32 + l15) * K + hl * 16;
  const u16* ap1 = ap0 + (size_t)16 * K;
  const u16* bp  = Bt + (size_t)(tn * 64 + l15) * K + hl * 16;
  size_t bs = (size_t)16 * K;
  v8f c00 = {}, c01 = {}, c02 = {}, c03 = {};
  v8f c10 = {}, c11 = {}, c12 = {}, c13 = {};
  for (int k0 = 0; k0 < K; k0 += 32) {
    // speculative prefetch of the next K-slab (harmless past the end)
    __builtin_prefetch((const void*)(ap0 + k0 + 256), 0, 0);
    __builtin_prefetch((const void*)(bp  + k0 + 256), 0, 0);
    v16bf a0 = ld_a(ap0 + k0);
    v16bf a1 = ld_a(ap1 + k0);
    v16bf b0 = ld_b(bp + k0);
    v16bf b1 = ld_b(bp + bs + k0);
    v16bf b2 = ld_b(bp + 2 * bs + k0);
    v16bf b3 = ld_b(bp + 3 * bs + k0);
    c00 = WMMA_BF16(a0, b0, c00);
    c10 = WMMA_BF16(a1, b0, c10);
    c01 = WMMA_BF16(a0, b1, c01);
    c11 = WMMA_BF16(a1, b1, c11);
    c02 = WMMA_BF16(a0, b2, c02);
    c12 = WMMA_BF16(a1, b2, c12);
    c03 = WMMA_BF16(a0, b3, c03);
    c13 = WMMA_BF16(a1, b3, c13);
  }
  float* cp0 = C + (size_t)(tm * 32 + hl * 8) * N + tn * 64 + l15;
  float* cp1 = cp0 + (size_t)16 * N;
#pragma unroll
  for (int r = 0; r < 8; ++r) {
    cp0[(size_t)r * N +  0] = c00[r];
    cp0[(size_t)r * N + 16] = c01[r];
    cp0[(size_t)r * N + 32] = c02[r];
    cp0[(size_t)r * N + 48] = c03[r];
    cp1[(size_t)r * N +  0] = c10[r];
    cp1[(size_t)r * N + 16] = c11[r];
    cp1[(size_t)r * N + 32] = c12[r];
    cp1[(size_t)r * N + 48] = c13[r];
  }
}

// ---------------- Q build: RoPE(geo) + fold exp(logit_scale)/sqrt(32), -> bf16 [b,h,t,64] ----------------
__global__ void k_build_q(const float* __restrict__ qs, const float* __restrict__ qg,
                          const float* __restrict__ ls, u16* __restrict__ Q) {
  int i = blockIdx.x * 256 + threadIdx.x;
  if (i >= kB * kT * kH * 64) return;
  int d = i & 63;
  int h = (i >> 6) & (kH - 1);
  int t = (i >> 10) & (kT - 1);
  int b = i >> 22;
  float scale = __expf(ls[h]) * 0.17677669529663687f;   // 1/sqrt(32)
  float val;
  if (d < kDS) {
    val = qs[((size_t)(b * kT + t) * kH + h) * kDS + d];
  } else {
    int g = d - kDS;
    int p = (g < 16) ? g : g - 16;
    float ang = (float)t * __powf(10000.0f, -(float)p * (1.0f / 16.0f));
    float sn, cs; __sincosf(ang, &sn, &cs);
    const float* bs = qg + ((size_t)(b * kT + t) * kH + h) * kDG;
    float a1 = bs[p], a2 = bs[p + 16];
    val = (g < 16) ? (a1 * cs - a2 * sn) : (a2 * cs + a1 * sn);
  }
  Q[((size_t)(b * kH + h) * kT + t) * 64 + d] = f2bf(val * scale);
}

// ---------------- KV build: RoPE(k_geo) then mean-pool; Kcat [b,kvh,key,64], Vt [b,kvh,64,key] ----------------
__global__ void k_build_kv(const float* __restrict__ ks, const float* __restrict__ kg,
                           const float* __restrict__ vv,
                           u16* __restrict__ Kc, u16* __restrict__ Vt) {
  int i = blockIdx.x * 256 + threadIdx.x;
  if (i >= kB * kHKV * kKP * 128) return;
  int d2 = i & 127;
  int j  = i >> 7;
  int key = j % kKP; j /= kKP;
  int kvh = j % kHKV;
  int b   = j / kHKV;
  bool isv = d2 >= 64;
  int d = isv ? d2 - 64 : d2;
  float outv = 0.0f;
  if (key < kKT) {
    if (isv) {
      if (key < kNB) {
        float s = 0.0f;
        for (int u = 0; u < kMB; ++u) {
          int t = key * kMB + u;
          s += vv[((size_t)(b * kT + t) * kHKV + kvh) * kDV + d];
        }
        outv = s * (1.0f / kMB);
      } else {
        int t = kRem + key - kNB;
        outv = vv[((size_t)(b * kT + t) * kHKV + kvh) * kDV + d];
      }
    } else if (d < kDS) {
      if (key < kNB) {
        float s = 0.0f;
        for (int u = 0; u < kMB; ++u) {
          int t = key * kMB + u;
          s += ks[((size_t)(b * kT + t) * kHKV + kvh) * kDS + d];
        }
        outv = s * (1.0f / kMB);
      } else {
        int t = kRem + key - kNB;
        outv = ks[((size_t)(b * kT + t) * kHKV + kvh) * kDS + d];
      }
    } else {
      int g = d - kDS;
      int p = (g < 16) ? g : g - 16;
      float invf = __powf(10000.0f, -(float)p * (1.0f / 16.0f));
      if (key < kNB) {
        float s = 0.0f;
        for (int u = 0; u < kMB; ++u) {
          int t = key * kMB + u;
          float sn, cs; __sincosf((float)t * invf, &sn, &cs);
          const float* bs = kg + ((size_t)(b * kT + t) * kHKV + kvh) * kDG;
          float a1 = bs[p], a2 = bs[p + 16];
          s += (g < 16) ? (a1 * cs - a2 * sn) : (a2 * cs + a1 * sn);
        }
        outv = s * (1.0f / kMB);
      } else {
        int t = kRem + key - kNB;
        float sn, cs; __sincosf((float)t * invf, &sn, &cs);
        const float* bs = kg + ((size_t)(b * kT + t) * kHKV + kvh) * kDG;
        float a1 = bs[p], a2 = bs[p + 16];
        outv = (g < 16) ? (a1 * cs - a2 * sn) : (a2 * cs + a1 * sn);
      }
    }
  }
  if (isv) Vt[((size_t)(b * kHKV + kvh) * 64 + d) * kKP + key] = f2bf(outv);
  else     Kc[((size_t)(b * kHKV + kvh) * kKP + key) * 64 + d] = f2bf(outv);
}

// ---------------- Flash attention, one wave per (b,h,16-query tile), S^T trick ----------------
__global__ void __launch_bounds__(256)
k_attn(const u16* __restrict__ Qc, const u16* __restrict__ Kc,
       const u16* __restrict__ Vt, u16* __restrict__ Ao) {
  int wid  = blockIdx.x * 8 + (threadIdx.x >> 5);
  int lane = threadIdx.x & 31;
  int t0  = (wid & 255) * 16;
  int h   = (wid >> 8) & (kH - 1);
  int b   = wid >> 12;
  int kvh = h >> 2;                      // H/HKV = 4
  int hl = lane >> 4, l15 = lane & 15, kb = hl * 8;

  // Q as B-operand for S^T = K * Q^T (N = query = lane&15, K-dim = feature d)
  const u16* qrow = Qc + ((size_t)(b * kH + h) * kT + t0 + l15) * 64 + hl * 16;
  v16bf q0 = ld_b(qrow);
  v16bf q1 = ld_b(qrow + 32);

  const u16* kcb = Kc + (size_t)(b * kHKV + kvh) * kKP * 64;
  const u16* vtb = Vt + (size_t)(b * kHKV + kvh) * 64 * kKP;

  v8f o0 = {}, o1 = {}, o2 = {}, o3 = {};    // O^T chunks [16d x 16q]
  float m = -INFINITY, l = 0.0f;
  int tq = t0 + l15;                          // this lane's query position

  for (int kt = 0; kt < kKP; kt += 32) {
    // K tiles as A-operand (M = key row = lane&15, K-dim = d)
    const u16* ka = kcb + (size_t)(kt + l15) * 64 + hl * 16;
    v16bf a00 = ld_a(ka);
    v16bf a01 = ld_a(ka + 32);
    v16bf a10 = ld_a(ka + 16 * 64);
    v16bf a11 = ld_a(ka + 16 * 64 + 32);
    v8f s0 = {}, s1 = {};
    s0 = WMMA_BF16(a00, q0, s0);
    s0 = WMMA_BF16(a01, q1, s0);
    s1 = WMMA_BF16(a10, q0, s1);
    s1 = WMMA_BF16(a11, q1, s1);

    float e0[8], e1[8];
    float tmax = -INFINITY;
#pragma unroll
    for (int r = 0; r < 8; ++r) {
      int key0 = kt + kb + r;
      int key1 = key0 + 16;
      float v0 = s0[r], v1 = s1[r];
      if (key0 >= kKT) v0 = -3.0e38f;
      else {
        int kp = (key0 < kNB) ? key0 * kMB + (kMB - 1) : kRem + key0 - kNB;
        if (kp > tq) v0 = -1.0e9f;           // reference's NEG_INF mask
      }
      if (key1 >= kKT) v1 = -3.0e38f;
      else {
        int kp = (key1 < kNB) ? key1 * kMB + (kMB - 1) : kRem + key1 - kNB;
        if (kp > tq) v1 = -1.0e9f;
      }
      e0[r] = v0; e1[r] = v1;
      tmax = fmaxf(tmax, fmaxf(v0, v1));
    }
    tmax = fmaxf(tmax, __shfl_xor(tmax, 16, 32));   // combine the two key-halves
    float mn   = fmaxf(m, tmax);
    float corr = __expf(m - mn);
    float tsum = 0.0f;
#pragma unroll
    for (int r = 0; r < 8; ++r) {
      e0[r] = __expf(e0[r] - mn);
      e1[r] = __expf(e1[r] - mn);
      tsum += e0[r] + e1[r];
    }
    tsum += __shfl_xor(tsum, 16, 32);
    l = l * corr + tsum;
    m = mn;
#pragma unroll
    for (int r = 0; r < 8; ++r) { o0[r] *= corr; o1[r] *= corr; o2[r] *= corr; o3[r] *= corr; }

    // P^T B-operand: hl=0 lane holds keys 0..15 = {local e0, partner e0};
    //                hl=1 lane holds keys 16..31 = {partner e1, local e1}.
    v16u pu;
#pragma unroll
    for (int r = 0; r < 8; ++r) {
      float p0 = __shfl_xor(e0[r], 16, 32);
      float p1 = __shfl_xor(e1[r], 16, 32);
      float lo = hl ? p1 : e0[r];
      float hi = hl ? e1[r] : p0;
      pu[r]     = f2bf(lo);
      pu[r + 8] = f2bf(hi);
    }
    v16bf pb = __builtin_bit_cast(v16bf, pu);

    // V^T chunks as A-operand (M = d row, K-dim = keys)
    const u16* va = vtb + (size_t)l15 * kKP + kt + hl * 16;
    v16bf w0 = ld_a(va);
    v16bf w1 = ld_a(va + 16 * kKP);
    v16bf w2 = ld_a(va + 32 * kKP);
    v16bf w3 = ld_a(va + 48 * kKP);
    o0 = WMMA_BF16(w0, pb, o0);
    o1 = WMMA_BF16(w1, pb, o1);
    o2 = WMMA_BF16(w2, pb, o2);
    o3 = WMMA_BF16(w3, pb, o3);
  }
  float rl = 1.0f / l;
  u16* ob = Ao + ((size_t)(b * kT + t0 + l15) * kH + h) * 64;   // [b,t,h,d] bf16
#pragma unroll
  for (int r = 0; r < 8; ++r) {
    int d = hl * 8 + r;
    ob[d]      = f2bf(o0[r] * rl);
    ob[d + 16] = f2bf(o1[r] * rl);
    ob[d + 32] = f2bf(o2[r] * rl);
    ob[d + 48] = f2bf(o3[r] * rl);
  }
}

// ---------------- host ----------------
extern "C" void kernel_launch(void* const* d_in, const int* /*in_sizes*/, int /*n_in*/,
                              void* d_out, int /*out_size*/, void* d_ws, size_t /*ws_size*/,
                              hipStream_t stream) {
  const float* x   = (const float*)d_in[0];
  const float* Wqs = (const float*)d_in[1];
  const float* Wks = (const float*)d_in[2];
  const float* Wqg = (const float*)d_in[3];
  const float* Wkg = (const float*)d_in[4];
  const float* Wv  = (const float*)d_in[5];
  const float* Wo  = (const float*)d_in[6];
  const float* ls  = (const float*)d_in[7];
  float* out = (float*)d_out;

  char* base = (char*)d_ws;
  size_t off = 0;
  auto carve = [&](size_t elems, size_t esz) -> void* {
    void* p = base + off;
    off += (elems * esz + 255) & ~(size_t)255;
    return p;
  };
  u16*   xb   = (u16*)  carve((size_t)kBT * kDM, 2);
  u16*   WqsT = (u16*)  carve((size_t)(kH * kDS)  * kDM, 2);
  u16*   WksT = (u16*)  carve((size_t)(kHKV * kDS) * kDM, 2);
  u16*   WqgT = (u16*)  carve((size_t)(kH * kDG)  * kDM, 2);
  u16*   WkgT = (u16*)  carve((size_t)(kHKV * kDG) * kDM, 2);
  u16*   WvT  = (u16*)  carve((size_t)(kHKV * kDV) * kDM, 2);
  u16*   WoT  = (u16*)  carve((size_t)kDM * (kH * kDV), 2);
  float* qse  = (float*)carve((size_t)kBT * kH * kDS, 4);
  float* kse  = (float*)carve((size_t)kBT * kHKV * kDS, 4);
  float* qge  = (float*)carve((size_t)kBT * kH * kDG, 4);
  float* kge  = (float*)carve((size_t)kBT * kHKV * kDG, 4);
  float* vbf  = (float*)carve((size_t)kBT * kHKV * kDV, 4);
  u16*   Qc   = (u16*)  carve((size_t)kB * kH * kT * 64, 2);
  u16*   Kc   = (u16*)  carve((size_t)kB * kHKV * kKP * 64, 2);
  u16*   Vt   = (u16*)  carve((size_t)kB * kHKV * 64 * kKP, 2);
  u16*   Ao   = (u16*)  carve((size_t)kBT * kH * kDV, 2);

  auto grid1 = [](int n) { return dim3((unsigned)((n + 255) / 256)); };

  { int n = kBT * kDM;
    k_cvt_bf16<<<grid1(n), 256, 0, stream>>>(x, xb, n); }
  k_transpose_bf16<<<grid1(kDM * kH * kDS),   256, 0, stream>>>(Wqs, WqsT, kDM, kH * kDS);
  k_transpose_bf16<<<grid1(kDM * kHKV * kDS), 256, 0, stream>>>(Wks, WksT, kDM, kHKV * kDS);
  k_transpose_bf16<<<grid1(kDM * kH * kDG),   256, 0, stream>>>(Wqg, WqgT, kDM, kH * kDG);
  k_transpose_bf16<<<grid1(kDM * kHKV * kDG), 256, 0, stream>>>(Wkg, WkgT, kDM, kHKV * kDG);
  k_transpose_bf16<<<grid1(kDM * kHKV * kDV), 256, 0, stream>>>(Wv,  WvT,  kDM, kHKV * kDV);
  k_transpose_bf16<<<grid1(kH * kDV * kDM),   256, 0, stream>>>(Wo,  WoT,  kH * kDV, kDM);

  auto gemm = [&](const u16* A, const u16* Bt, float* C, int M, int N, int K) {
    int tiles = (M / 32) * (N / 64);
    k_gemm_bf16<<<dim3((unsigned)((tiles + 7) / 8)), 256, 0, stream>>>(A, Bt, C, M, N, K);
  };
  gemm(xb, WqsT, qse, kBT, kH * kDS,   kDM);
  gemm(xb, WksT, kse, kBT, kHKV * kDS, kDM);
  gemm(xb, WqgT, qge, kBT, kH * kDG,   kDM);
  gemm(xb, WkgT, kge, kBT, kHKV * kDG, kDM);
  gemm(xb, WvT,  vbf, kBT, kHKV * kDV, kDM);

  k_build_q <<<grid1(kB * kT * kH * 64),     256, 0, stream>>>(qse, qge, ls, Qc);
  k_build_kv<<<grid1(kB * kHKV * kKP * 128), 256, 0, stream>>>(kse, kge, vbf, Kc, Vt);

  { int waves = kB * kH * (kT / 16);
    k_attn<<<dim3((unsigned)(waves / 8)), 256, 0, stream>>>(Qc, Kc, Vt, Ao); }

  gemm(Ao, WoT, out, kBT, kDM, kH * kDV);
}
